// EfficientFaceTransformer_89206470738428
// MI455X (gfx1250) — matrix operand order repaired and checked by reading, not compiled, not run on hardware
//
#include <hip/hip_runtime.h>
#include <hip/hip_bf16.h>

typedef __bf16 bf16;
typedef bf16 v16bf __attribute__((ext_vector_type(16)));
typedef bf16 v8bf  __attribute__((ext_vector_type(8)));
typedef float v8f  __attribute__((ext_vector_type(8)));

#define DEV static __device__ __forceinline__

// ---------------- wave reductions (wave32) ----------------
DEV float wred_sum(float v) {
#pragma unroll
  for (int o = 16; o > 0; o >>= 1) v += __shfl_xor(v, o);
  return v;
}
DEV float wred_max(float v) {
#pragma unroll
  for (int o = 16; o > 0; o >>= 1) v = fmaxf(v, __shfl_xor(v, o));
  return v;
}

// ---------------- CDNA5 async copy (global -> LDS), ASYNCcnt tracked ----------------
DEV void async_ld16(unsigned lds_byte_off, const bf16* g) {
  asm volatile("global_load_async_to_lds_b128 %0, %1, off"
               :
               : "v"(lds_byte_off), "v"(g)
               : "memory");
}
DEV void wait_async_le4() { asm volatile("s_wait_asynccnt 0x4" ::: "memory"); }
DEV void wait_async_0()   { asm volatile("s_wait_asynccnt 0x0" ::: "memory"); }
DEV unsigned lds_off32(const void* p) { return (unsigned)(uintptr_t)p; }  // LDS offset: low 32 bits of the address

// ---------------- WMMA fragment loaders (bf16, 16x16x32) ----------------
// A 16x32 (MxK): lanes 0-15 row M=lane, elems 0-7 -> K=0..7, 8-15 -> K=16..23
//                lanes 16-31 row M=lane-16, elems 0-7 -> K=8..15, 8-15 -> K=24..31
DEV v16bf frag_a(const bf16* __restrict__ base, int lane, int ld) {
  const int m = lane & 15;
  const int klo = (lane & 16) ? 8 : 0;
  const bf16* p = base + (size_t)m * ld + klo;
  v8bf lo = *(const v8bf*)(p);
  v8bf hi = *(const v8bf*)(p + 16);
  v16bf r;
#pragma unroll
  for (int i = 0; i < 8; ++i) { r[i] = lo[i]; r[8 + i] = hi[i]; }
  return r;
}
// B 32x16 (KxN), sourced from row-major W[N,K]: lane column n=lane&15,
// lanes 0-15 hold K=0..7,16..23; lanes 16-31 hold K=8..15,24..31 (matches frag_a).
DEV v16bf frag_b(const bf16* __restrict__ base, int lane, int ld) {
  const int n = lane & 15;
  const int klo = (lane & 16) ? 8 : 0;
  const bf16* p = base + (size_t)n * ld + klo;
  v8bf lo = *(const v8bf*)(p);
  v8bf hi = *(const v8bf*)(p + 16);
  v16bf r;
#pragma unroll
  for (int i = 0; i < 8; ++i) { r[i] = lo[i]; r[8 + i] = hi[i]; }
  return r;
}
DEV v8f wmma_bf16(v16bf a, v16bf b, v8f c) {
  return __builtin_amdgcn_wmma_f32_16x16x32_bf16(false, a, false, b, (short)0, c,
                                                 false, false);
}

// ---------------- layout transposes ----------------
__global__ void transpose_in_kernel(const float* __restrict__ x, float* __restrict__ xt) {
  int idx = blockIdx.x * 256 + threadIdx.x;  // over B*512*784
  if (idx >= 16 * 512 * 784) return;
  int n = idx % 784;
  int bc = idx / 784;
  int c = bc & 511;
  int b = bc >> 9;
  xt[((size_t)(b * 784 + n)) * 512 + c] = x[idx];
}
__global__ void transpose_out_kernel(const float* __restrict__ f, float* __restrict__ out) {
  int idx = blockIdx.x * 256 + threadIdx.x;  // over B*512*784 (NCHW order)
  if (idx >= 16 * 512 * 784) return;
  int n = idx % 784;
  int bc = idx / 784;
  int c = bc & 511;
  int b = bc >> 9;
  out[idx] = f[((size_t)(b * 784 + n)) * 512 + c];
}

// ---------------- pooling over tokens (token-major) ----------------
__global__ void pool_kernel(const float* __restrict__ X, float* __restrict__ pooled) {
  int i = blockIdx.x * 256 + threadIdx.x;  // over 16*512
  if (i >= 16 * 512) return;
  int b = i >> 9, c = i & 511;
  const float* p = X + (size_t)b * 784 * 512 + c;
  float s = 0.0f;
  for (int n = 0; n < 784; ++n) s += p[(size_t)n * 512];
  pooled[i] = s * (1.0f / 784.0f);
}

// ---------------- adaptive-LN parameter MLP (per batch) ----------------
__global__ void adapt_kernel(const float* __restrict__ pooled,
                             const float* __restrict__ a1w, const float* __restrict__ a1b,
                             const float* __restrict__ a2w, const float* __restrict__ a2b,
                             float* __restrict__ ad) {
  int b = blockIdx.x, t = threadIdx.x;  // 256 threads
  __shared__ float sp[512];
  __shared__ float sh[128];
  for (int c = t; c < 512; c += 256) sp[c] = pooled[b * 512 + c];
  __syncthreads();
  if (t < 128) {
    float acc = a1b[t];
    for (int k = 0; k < 512; ++k) acc += a1w[t * 512 + k] * sp[k];
    sh[t] = fmaxf(acc, 0.0f);
  }
  __syncthreads();
  for (int o = t; o < 1024; o += 256) {
    float acc = a2b[o];
    for (int k = 0; k < 128; ++k) acc += a2w[o * 128 + k] * sh[k];
    ad[b * 1024 + o] = acc;
  }
}

// ---------------- adaptive LayerNorm over channels (wave per token) ----------------
__global__ void __launch_bounds__(32)
ln_kernel(const float* __restrict__ X, const float* __restrict__ w,
          const float* __restrict__ bb, const float* __restrict__ ad,
          bf16* __restrict__ outb, float* __restrict__ outf) {
  int m = blockIdx.x, lane = threadIdx.x;
  const float* row = X + (size_t)m * 512;
  float ls = 0.0f;
  for (int c = lane; c < 512; c += 32) ls += row[c];
  float u = wred_sum(ls) * (1.0f / 512.0f);
  float lv = 0.0f;
  for (int c = lane; c < 512; c += 32) { float d = row[c] - u; lv += d * d; }
  float var = wred_sum(lv) * (1.0f / 512.0f);
  float inv = rsqrtf(var + 1e-6f);
  int b = m / 784;
  const float* aw = ad + (size_t)b * 1024;
  const float* ab = aw + 512;
  for (int c = lane; c < 512; c += 32) {
    float xn = (row[c] - u) * inv;
    float y = w[c] * (1.0f + aw[c]) * xn + (bb[c] + ab[c]);
    outb[(size_t)m * 512 + c] = (bf16)y;
    if (outf) outf[(size_t)m * 512 + c] = y;
  }
}

// ---------------- generic conversions ----------------
__global__ void cvt_pad_bf_kernel(bf16* __restrict__ dst, const float* __restrict__ src,
                                  int R, int Cd, int Cs, int SR) {
  int i = blockIdx.x * 256 + threadIdx.x;
  if (i >= R * Cd) return;
  int r = i / Cd, c = i % Cd;
  dst[i] = (r < SR && c < Cs) ? (bf16)src[(size_t)r * Cs + c] : (bf16)0.0f;
}
__global__ void pad_bias_kernel(float* __restrict__ dst, const float* __restrict__ src,
                                int n, int ns) {
  int i = blockIdx.x * 256 + threadIdx.x;
  if (i < n) dst[i] = (i < ns) ? src[i] : 0.0f;
}

// ---------------- WMMA GEMM v2: 64x64 block tile, 4 waves, async LDS double buffer --------
// C[M,N] = A[M,K] * W[N,K]^T (+ epilogue). Requires M%64==0, N%64==0, K%32==0.
// mode 0: Cf = acc+bias            mode 1: Cf = acc+bias+res
// mode 2: Cb = gelu(acc+bias)      mode 3: Cf += rsc[b*3]*(acc+bias)
// mode 4: Cf = res + rsc[b*3]*(acc+bias)   mode 5: Cb = relu(acc+bias)
__global__ void __launch_bounds__(128)
gemm_wmma_kernel(const bf16* __restrict__ A, const bf16* __restrict__ W,
                 const float* __restrict__ bias, float* __restrict__ Cf,
                 bf16* __restrict__ Cb, const float* __restrict__ res,
                 const float* __restrict__ rsc, int K, int ldc, int mode) {
  __shared__ __align__(16) bf16 sA[2][64 * 32];
  __shared__ __align__(16) bf16 sB[2][64 * 32];
  const int tid = threadIdx.x;
  const int lane = tid & 31;
  const int wv = tid >> 5;  // wave 0..3 -> M sub-tile
  const int n0 = blockIdx.x * 64;
  const int m0 = blockIdx.y * 64;
  const bf16* Ab = A + (size_t)m0 * K;
  const bf16* Wb = W + (size_t)n0 * K;

  // each thread stages 32B of A and 32B of B per k-step (2x b128 each)
  const int trow = tid >> 1;          // 0..63
  const int tce = (tid & 1) * 16;     // bf16 element col: 0 or 16
  const unsigned oA[2] = {lds_off32(&sA[0][trow * 32 + tce]),
                          lds_off32(&sA[1][trow * 32 + tce])};
  const unsigned oB[2] = {lds_off32(&sB[0][trow * 32 + tce]),
                          lds_off32(&sB[1][trow * 32 + tce])};

  auto issue = [&](int buf, int kk) {
    const bf16* ga = Ab + (size_t)trow * K + kk + tce;
    const bf16* gb = Wb + (size_t)trow * K + kk + tce;
    async_ld16(oA[buf], ga);
    async_ld16(oA[buf] + 16, ga + 8);
    async_ld16(oB[buf], gb);
    async_ld16(oB[buf] + 16, gb + 8);
  };

  const int nsteps = K >> 5;
  issue(0, 0);
  v8f acc0 = {}, acc1 = {}, acc2 = {}, acc3 = {};
  for (int s = 0; s < nsteps; ++s) {
    const int buf = s & 1;
    if (s + 1 < nsteps) {
      issue(buf ^ 1, (s + 1) << 5);
      wait_async_le4();  // stage s resident; stage s+1 in flight
    } else {
      wait_async_0();
    }
    __syncthreads();
    v16bf a = frag_a(&sA[buf][(wv * 16) * 32], lane, 32);
    acc0 = wmma_bf16(a, frag_b(&sB[buf][0], lane, 32), acc0);
    acc1 = wmma_bf16(a, frag_b(&sB[buf][16 * 32], lane, 32), acc1);
    acc2 = wmma_bf16(a, frag_b(&sB[buf][32 * 32], lane, 32), acc2);
    acc3 = wmma_bf16(a, frag_b(&sB[buf][48 * 32], lane, 32), acc3);
    __syncthreads();  // protect buf from next overwrite
  }

  const int mbase = m0 + wv * 16 + ((lane & 16) ? 8 : 0);
  const float* accs[4] = {(const float*)&acc0, (const float*)&acc1,
                          (const float*)&acc2, (const float*)&acc3};
#pragma unroll
  for (int nt = 0; nt < 4; ++nt) {
    const int n = n0 + nt * 16 + (lane & 15);
    const float bv = bias ? bias[n] : 0.0f;
#pragma unroll
    for (int r = 0; r < 8; ++r) {
      const int m = mbase + r;
      float v = accs[nt][r] + bv;
      size_t o = (size_t)m * ldc + n;
      if (mode == 0)      Cf[o] = v;
      else if (mode == 1) Cf[o] = v + res[o];
      else if (mode == 2) Cb[o] = (bf16)(0.5f * v * (1.0f + erff(v * 0.70710678118f)));
      else if (mode == 3) Cf[o] += rsc[(m / 784) * 3] * v;
      else if (mode == 4) Cf[o] = res[o] + rsc[(m / 784) * 3] * v;
      else                Cb[o] = (bf16)fmaxf(v, 0.0f);
    }
  }
}

// ---------------- qkv repack: f32 [B*N,1536] -> q/k bf16 [B,H,N,64], vT bf16 [B,H,64,800] --------
__global__ void repack_qkv_kernel(const float* __restrict__ qkv, bf16* __restrict__ q,
                                  bf16* __restrict__ k, bf16* __restrict__ vT) {
  int idx = blockIdx.x * 256 + threadIdx.x;  // over 16*8*784*64
  if (idx >= 16 * 8 * 784 * 64) return;
  int d = idx & 63;
  int t = idx >> 6;
  int n = t % 784;
  t /= 784;
  int h = t & 7;
  int b = t >> 3;
  size_t src = (size_t)(b * 784 + n) * 1536 + h * 64 + d;
  q[idx] = (bf16)(qkv[src] * 0.125f);  // fold HD^-0.5
  k[idx] = (bf16)qkv[src + 512];
  vT[((size_t)((b * 8 + h) * 64 + d)) * 800 + n] = (bf16)qkv[src + 1024];
}
__global__ void vt_pad_kernel(bf16* __restrict__ vT) {
  int i = blockIdx.x * 256 + threadIdx.x;  // over 16*8*64*16
  if (i >= 16 * 8 * 64 * 16) return;
  int t = i & 15;
  int row = i >> 4;
  vT[(size_t)row * 800 + 784 + t] = (bf16)0.0f;
}

// ---------------- attention: wave per (16-query tile, head, batch) ----------------
__global__ void __launch_bounds__(32)
attn_kernel(const bf16* __restrict__ qb, const bf16* __restrict__ kb,
            const bf16* __restrict__ vT, const float* __restrict__ SP,
            const float* __restrict__ FP, const float* __restrict__ RPB,
            bf16* __restrict__ ob) {
  const int qt = blockIdx.x, h = blockIdx.y, b = blockIdx.z;
  const int lane = threadIdx.x;
  __shared__ float sS[16][800];
  const int bh = b * 8 + h;
  const bf16* qbase = qb + ((size_t)bh * 784 + qt * 16) * 64;
  const v16bf a0 = frag_a(qbase, lane, 64);
  const v16bf a1 = frag_a(qbase + 32, lane, 64);
  const int n = lane & 15;
  const int rbase = (lane & 16) ? 8 : 0;

  // --- scores: QK^T + struct_prior + rpb, * face_prior ---
  for (int t = 0; t < 49; ++t) {
    const bf16* kbase = kb + ((size_t)bh * 784 + t * 16) * 64;
    v8f acc = {};
    acc = wmma_bf16(a0, frag_b(kbase, lane, 64), acc);
    acc = wmma_bf16(a1, frag_b(kbase + 32, lane, 64), acc);
    const int key = t * 16 + n;
    const int y2 = key / 28, x2 = key % 28;
#pragma unroll
    for (int r = 0; r < 8; ++r) {
      const int rl = rbase + r;
      const int qr = qt * 16 + rl;
      const size_t o = (size_t)qr * 784 + key;
      const float sp = SP[(size_t)h * 614656 + o];
      const float fp = FP[(size_t)b * 614656 + o];
      const int y1 = qr / 28, x1 = qr % 28;
      const int ridx = (y1 - y2 + 27) * 55 + (x1 - x2 + 27);
      const float rp = RPB[ridx * 8 + h];
      sS[rl][key] = (acc[r] + sp + rp) * fp;
    }
  }
  __syncthreads();

  // --- softmax per row over 784 keys ---
  for (int r = 0; r < 16; ++r) {
    float lm = -1e30f;
    for (int m = lane; m < 784; m += 32) lm = fmaxf(lm, sS[r][m]);
    lm = wred_max(lm);
    float ls = 0.0f;
    for (int m = lane; m < 784; m += 32) {
      float p = expf(sS[r][m] - lm);
      sS[r][m] = p;
      ls += p;
    }
    ls = wred_sum(ls);
    float inv = 1.0f / ls;
    for (int m = lane; m < 784; m += 32) sS[r][m] *= inv;
    if (lane < 16) sS[r][784 + lane] = 0.0f;  // zero pad keys 784..799
  }
  __syncthreads();

  // --- PV: O[16,64] = P[16,784] * V[784,64] (V stored d-major) ---
  const bf16* vbase = vT + (size_t)bh * 64 * 800;
#pragma unroll
  for (int dt = 0; dt < 4; ++dt) {
    v8f acc = {};
    for (int ks = 0; ks < 25; ++ks) {
      v16bf ap;
      {
        const int m2 = lane & 15;
        const int klo = (lane & 16) ? 8 : 0;
        const float* p = &sS[m2][ks * 32 + klo];
#pragma unroll
        for (int i = 0; i < 8; ++i) { ap[i] = (bf16)p[i]; ap[8 + i] = (bf16)p[i + 16]; }
      }
      v16bf bp = frag_b(vbase + (size_t)(dt * 16) * 800 + ks * 32, lane, 800);
      acc = wmma_bf16(ap, bp, acc);
    }
#pragma unroll
    for (int r = 0; r < 8; ++r) {
      const int qr = qt * 16 + rbase + r;
      ob[((size_t)(b * 784 + qr)) * 512 + h * 64 + dt * 16 + n] = (bf16)acc[r];
    }
  }
}

// ---------------- depthwise convs (3/5/7) * sigmoid(gate) -> bf16 ----------------
__global__ void dwgate_kernel(const float* __restrict__ xn2f, const float* __restrict__ ew,
                              const float* __restrict__ eb, const float* __restrict__ nw,
                              const float* __restrict__ nb, const float* __restrict__ mw,
                              const float* __restrict__ mb, const float* __restrict__ gl,
                              bf16* __restrict__ catg) {
  int idx = blockIdx.x * 256 + threadIdx.x;  // over 12544*1536
  if (idx >= 12544 * 1536) return;
  int j = idx % 1536;
  int m = idx / 1536;
  int b = m / 784, nn = m % 784;
  int y = nn / 28, x = nn % 28;
  int g = j >> 9, c = j & 511;
  int ks;
  const float* wp;
  const float* bp;
  if (g == 0) { ks = 3; wp = ew + c * 9; bp = eb; }
  else if (g == 1) { ks = 5; wp = nw + c * 25; bp = nb; }
  else { ks = 7; wp = mw + c * 49; bp = mb; }
  int p = ks >> 1;
  float acc = bp[c];
  for (int dy = -p; dy <= p; ++dy) {
    int yy = y + dy;
    if (yy < 0 || yy >= 28) continue;
    for (int dx = -p; dx <= p; ++dx) {
      int xx = x + dx;
      if (xx < 0 || xx >= 28) continue;
      acc += xn2f[((size_t)(b * 784 + yy * 28 + xx)) * 512 + c] * wp[(dy + p) * ks + (dx + p)];
    }
  }
  float gv = 1.0f / (1.0f + expf(-gl[(size_t)m * 1536 + j]));
  catg[(size_t)m * 1536 + j] = (bf16)(acc * gv);
}

// ---------------- router softmax ----------------
__global__ void __launch_bounds__(32)
router_kernel(const float* __restrict__ pooled, const float* __restrict__ rw,
              const float* __restrict__ rb, float* __restrict__ r) {
  int b = blockIdx.x, lane = threadIdx.x;
  float d0 = 0, d1 = 0, d2 = 0;
  for (int c = lane; c < 512; c += 32) {
    float pv = pooled[b * 512 + c];
    d0 += pv * rw[c];
    d1 += pv * rw[512 + c];
    d2 += pv * rw[1024 + c];
  }
  d0 = wred_sum(d0);
  d1 = wred_sum(d1);
  d2 = wred_sum(d2);
  if (lane == 0) {
    float l0 = d0 + rb[0], l1 = d1 + rb[1], l2 = d2 + rb[2];
    float mx = fmaxf(l0, fmaxf(l1, l2));
    float e0 = expf(l0 - mx), e1 = expf(l1 - mx), e2 = expf(l2 - mx);
    float s = 1.0f / (e0 + e1 + e2);
    r[b * 3 + 0] = e0 * s;
    r[b * 3 + 1] = e1 * s;
    r[b * 3 + 2] = e2 * s;
  }
}

// =====================================================================
extern "C" void kernel_launch(void* const* d_in, const int* in_sizes, int n_in,
                              void* d_out, int out_size, void* d_ws, size_t ws_size,
                              hipStream_t stream) {
  (void)in_sizes; (void)n_in; (void)out_size; (void)ws_size;
  const float* x_in = (const float*)d_in[0];
  const float* face = (const float*)d_in[1];
  const float* n1_w = (const float*)d_in[2];
  const float* n1_b = (const float*)d_in[3];
  const float* n1_a1w = (const float*)d_in[4];
  const float* n1_a1b = (const float*)d_in[5];
  const float* n1_a2w = (const float*)d_in[6];
  const float* n1_a2b = (const float*)d_in[7];
  const float* n2_w = (const float*)d_in[8];
  const float* n2_b = (const float*)d_in[9];
  const float* n2_a1w = (const float*)d_in[10];
  const float* n2_a1b = (const float*)d_in[11];
  const float* n2_a2w = (const float*)d_in[12];
  const float* n2_a2b = (const float*)d_in[13];
  const float* qkv_w = (const float*)d_in[14];
  const float* qkv_b = (const float*)d_in[15];
  const float* rpb = (const float*)d_in[16];
  const float* sprior = (const float*)d_in[17];
  const float* proj_w = (const float*)d_in[18];
  const float* proj_b = (const float*)d_in[19];
  const float* eye_w = (const float*)d_in[20];
  const float* eye_b = (const float*)d_in[21];
  const float* nose_w = (const float*)d_in[22];
  const float* nose_b = (const float*)d_in[23];
  const float* mouth_w = (const float*)d_in[24];
  const float* mouth_b = (const float*)d_in[25];
  const float* fusion_w = (const float*)d_in[26];
  const float* fusion_b = (const float*)d_in[27];
  const float* gate1_w = (const float*)d_in[28];
  const float* gate1_b = (const float*)d_in[29];
  const float* gate2_w = (const float*)d_in[30];
  const float* gate2_b = (const float*)d_in[31];
  const float* e1w[3] = {(const float*)d_in[32], (const float*)d_in[36], (const float*)d_in[40]};
  const float* e1b[3] = {(const float*)d_in[33], (const float*)d_in[37], (const float*)d_in[41]};
  const float* e2w[3] = {(const float*)d_in[34], (const float*)d_in[38], (const float*)d_in[42]};
  const float* e2b[3] = {(const float*)d_in[35], (const float*)d_in[39], (const float*)d_in[43]};
  const float* router_w = (const float*)d_in[44];
  const float* router_b = (const float*)d_in[45];
  float* out = (float*)d_out;

  // ---------------- workspace arena ----------------
  constexpr size_t MT = 12544;           // tokens
  constexpr size_t SZ_XF = MT * 512 * 4; // 25,690,112
  char* ws = (char*)d_ws;
  constexpr size_t O_XT = 0;
  constexpr size_t O_X1 = O_XT + SZ_XF;
  constexpr size_t O_X2 = O_X1 + SZ_XF;
  constexpr size_t O_XN2F = O_X2 + SZ_XF;
  constexpr size_t O_FIN = O_XN2F + SZ_XF;
  constexpr size_t O_ABF = O_FIN + SZ_XF;            // bf16 activations (reused)
  constexpr size_t O_QKVF = O_ABF + MT * 512 * 2;    // qkv f32 / gate logits
  constexpr size_t O_R1 = O_QKVF + MT * 1536 * 4;    // q/k/vT/o ; reused: catg, mh
  constexpr size_t O_QB = O_R1;
  constexpr size_t O_KB = O_QB + MT * 512 * 2;
  constexpr size_t O_VT = O_KB + MT * 512 * 2;
  constexpr size_t SZ_VT = (size_t)16 * 8 * 64 * 800 * 2;
  constexpr size_t O_OB = O_VT + SZ_VT;
  constexpr size_t O_CATG = O_R1;                    // 12544*1536*2, after attn dead
  constexpr size_t O_MH = O_R1;                      // 12544*704*2, after fusion dead
  constexpr size_t O_GH = O_OB + MT * 512 * 2;
  constexpr size_t O_W = O_GH + MT * 128 * 2;
  constexpr size_t O_Wqkv = O_W;
  constexpr size_t O_Wproj = O_Wqkv + (size_t)1536 * 512 * 2;
  constexpr size_t O_Wfus = O_Wproj + (size_t)512 * 512 * 2;
  constexpr size_t O_Wg1 = O_Wfus + (size_t)512 * 1536 * 2;
  constexpr size_t O_Wg2 = O_Wg1 + (size_t)128 * 512 * 2;
  constexpr size_t O_We1 = O_Wg2 + (size_t)1536 * 128 * 2;   // 3 x [704,512]
  constexpr size_t O_We2 = O_We1 + (size_t)3 * 704 * 512 * 2; // 3 x [512,704]
  constexpr size_t O_B1 = O_We2 + (size_t)3 * 512 * 704 * 2;  // 3 x 704 f32
  constexpr size_t O_POOL = O_B1 + (size_t)3 * 704 * 4;
  constexpr size_t O_AD1 = O_POOL + 16 * 512 * 4;
  constexpr size_t O_AD2 = O_AD1 + 16 * 1024 * 4;
  constexpr size_t O_RR = O_AD2 + 16 * 1024 * 4;

  float* xt = (float*)(ws + O_XT);
  float* x1 = (float*)(ws + O_X1);
  float* x2 = (float*)(ws + O_X2);
  float* xn2f = (float*)(ws + O_XN2F);
  float* fin = (float*)(ws + O_FIN);
  bf16* abf = (bf16*)(ws + O_ABF);
  float* qkvf = (float*)(ws + O_QKVF);
  float* glogit = (float*)(ws + O_QKVF);
  bf16* q_bf = (bf16*)(ws + O_QB);
  bf16* k_bf = (bf16*)(ws + O_KB);
  bf16* vT_bf = (bf16*)(ws + O_VT);
  bf16* o_bf = (bf16*)(ws + O_OB);
  bf16* catg = (bf16*)(ws + O_CATG);
  bf16* mh = (bf16*)(ws + O_MH);
  bf16* gh = (bf16*)(ws + O_GH);
  bf16* wqkv = (bf16*)(ws + O_Wqkv);
  bf16* wproj = (bf16*)(ws + O_Wproj);
  bf16* wfus = (bf16*)(ws + O_Wfus);
  bf16* wg1 = (bf16*)(ws + O_Wg1);
  bf16* wg2 = (bf16*)(ws + O_Wg2);
  float* pooled = (float*)(ws + O_POOL);
  float* ad1 = (float*)(ws + O_AD1);
  float* ad2 = (float*)(ws + O_AD2);
  float* rbuf = (float*)(ws + O_RR);

  auto cdiv = [](int a, int b) { return (a + b - 1) / b; };

  // 1) transpose input to token-major
  transpose_in_kernel<<<cdiv(16 * 512 * 784, 256), 256, 0, stream>>>(x_in, xt);

  // 2) weight conversions (f32 -> bf16, with padding for MoE)
  cvt_pad_bf_kernel<<<cdiv(1536 * 512, 256), 256, 0, stream>>>(wqkv, qkv_w, 1536, 512, 512, 1536);
  cvt_pad_bf_kernel<<<cdiv(512 * 512, 256), 256, 0, stream>>>(wproj, proj_w, 512, 512, 512, 512);
  cvt_pad_bf_kernel<<<cdiv(512 * 1536, 256), 256, 0, stream>>>(wfus, fusion_w, 512, 1536, 1536, 512);
  cvt_pad_bf_kernel<<<cdiv(128 * 512, 256), 256, 0, stream>>>(wg1, gate1_w, 128, 512, 512, 128);
  cvt_pad_bf_kernel<<<cdiv(1536 * 128, 256), 256, 0, stream>>>(wg2, gate2_w, 1536, 128, 128, 1536);
  for (int e = 0; e < 3; ++e) {
    bf16* w1 = (bf16*)(ws + O_We1 + (size_t)e * 704 * 512 * 2);
    bf16* w2 = (bf16*)(ws + O_We2 + (size_t)e * 512 * 704 * 2);
    float* b1 = (float*)(ws + O_B1 + (size_t)e * 704 * 4);
    cvt_pad_bf_kernel<<<cdiv(704 * 512, 256), 256, 0, stream>>>(w1, e1w[e], 704, 512, 512, 682);
    cvt_pad_bf_kernel<<<cdiv(512 * 704, 256), 256, 0, stream>>>(w2, e2w[e], 512, 704, 682, 512);
    pad_bias_kernel<<<cdiv(704, 256), 256, 0, stream>>>(b1, e1b[e], 704, 682);
  }

  // 3) adaptive LN #1
  pool_kernel<<<32, 256, 0, stream>>>(xt, pooled);
  adapt_kernel<<<16, 256, 0, stream>>>(pooled, n1_a1w, n1_a1b, n1_a2w, n1_a2b, ad1);
  ln_kernel<<<12544, 32, 0, stream>>>(xt, n1_w, n1_b, ad1, abf, nullptr);

  // 4) QKV GEMM + repack
  gemm_wmma_kernel<<<dim3(24, 196), 128, 0, stream>>>(abf, wqkv, qkv_b, qkvf, nullptr,
                                                      nullptr, nullptr, 512, 1536, 0);
  repack_qkv_kernel<<<cdiv(16 * 8 * 784 * 64, 256), 256, 0, stream>>>(qkvf, q_bf, k_bf, vT_bf);
  vt_pad_kernel<<<cdiv(16 * 8 * 64 * 16, 256), 256, 0, stream>>>(vT_bf);

  // 5) attention (fused biases + softmax + PV)
  attn_kernel<<<dim3(49, 8, 16), 32, 0, stream>>>(q_bf, k_bf, vT_bf, sprior, face, rpb, o_bf);

  // 6) proj GEMM + residual
  gemm_wmma_kernel<<<dim3(8, 196), 128, 0, stream>>>(o_bf, wproj, proj_b, x1, nullptr,
                                                     xt, nullptr, 512, 512, 1);

  // 7) adaptive LN #2 (keeps f32 copy for depthwise convs)
  pool_kernel<<<32, 256, 0, stream>>>(x1, pooled);
  adapt_kernel<<<16, 256, 0, stream>>>(pooled, n2_a1w, n2_a1b, n2_a2w, n2_a2b, ad2);
  ln_kernel<<<12544, 32, 0, stream>>>(x1, n2_w, n2_b, ad2, abf, xn2f);

  // 8) gate: relu(1x1) -> 1x1 (sigmoid fused in dwgate)
  gemm_wmma_kernel<<<dim3(2, 196), 128, 0, stream>>>(abf, wg1, gate1_b, nullptr, gh,
                                                     nullptr, nullptr, 512, 128, 5);
  gemm_wmma_kernel<<<dim3(24, 196), 128, 0, stream>>>(gh, wg2, gate2_b, glogit, nullptr,
                                                      nullptr, nullptr, 128, 1536, 0);

  // 9) depthwise 3/5/7 convs * sigmoid(gate) -> bf16 cat
  dwgate_kernel<<<cdiv(12544 * 1536, 256), 256, 0, stream>>>(
      xn2f, eye_w, eye_b, nose_w, nose_b, mouth_w, mouth_b, glogit, catg);

  // 10) fusion GEMM + residual -> x2
  gemm_wmma_kernel<<<dim3(8, 196), 128, 0, stream>>>(catg, wfus, fusion_b, x2, nullptr,
                                                     x1, nullptr, 1536, 512, 1);

  // 11) router
  pool_kernel<<<32, 256, 0, stream>>>(x2, pooled);
  router_kernel<<<16, 32, 0, stream>>>(pooled, router_w, router_b, rbuf);

  // 12) MoE experts: fin = x2 + sum_e r_e * (W2 gelu(W1 x2 + b1) + b2)
  cvt_pad_bf_kernel<<<cdiv(12544 * 512, 256), 256, 0, stream>>>(abf, x2, 12544, 512, 512, 12544);
  for (int e = 0; e < 3; ++e) {
    bf16* w1 = (bf16*)(ws + O_We1 + (size_t)e * 704 * 512 * 2);
    bf16* w2 = (bf16*)(ws + O_We2 + (size_t)e * 512 * 704 * 2);
    float* b1 = (float*)(ws + O_B1 + (size_t)e * 704 * 4);
    gemm_wmma_kernel<<<dim3(11, 196), 128, 0, stream>>>(abf, w1, b1, nullptr, mh,
                                                        nullptr, nullptr, 512, 704, 2);
    gemm_wmma_kernel<<<dim3(8, 196), 128, 0, stream>>>(mh, w2, e2b[e], fin, nullptr,
                                                       x2, rbuf + e, 704, 512, e == 0 ? 4 : 3);
  }

  // 13) transpose back to NCHW
  transpose_out_kernel<<<cdiv(16 * 512 * 784, 256), 256, 0, stream>>>(fin, out);
}
